// EncoderLayer_18966575579702
// MI455X (gfx1250) — matrix operand order repaired and checked by reading, not compiled, not run on hardware
//
#include <hip/hip_runtime.h>
#include <stdint.h>

#define BB 16
#define SS 512
#define DD 1024
#define HH 16
#define HDIM 64
#define DHALF 32
#define DFFN 4096
#define BSROWS (BB*SS)

typedef __attribute__((ext_vector_type(16))) __bf16 v16bf;
typedef __attribute__((ext_vector_type(8)))  float  v8f;

// ---------- bf16 helpers (RNE, bit-exact storage) ----------
static __device__ __forceinline__ unsigned short f2bf_bits(float f) {
  union { float f; uint32_t u; } c; c.f = f;
  uint32_t u = c.u;
  u = u + 0x7FFFu + ((u >> 16) & 1u);
  return (unsigned short)(u >> 16);
}
static __device__ __forceinline__ __bf16 f2bf(float f) {
  union { unsigned short s; __bf16 h; } c; c.s = f2bf_bits(f); return c.h;
}
static __device__ __forceinline__ float bf2f(__bf16 h) {
  union { __bf16 h; unsigned short s; } c; c.h = h;
  union { uint32_t u; float f; } d; d.u = ((uint32_t)c.s) << 16; return d.f;
}

// ---------- WMMA fragment loaders (ISA 7.12.2 layouts, wave32) ----------
// A (16x32 bf16, MxK): lane<16 holds M=lane, K=0..7 (elems 0..7) and K=16..23
// (elems 8..15); lane>=16 holds K=8..15 / K=24..31.
static __device__ __forceinline__ v16bf load_frag_a(const __bf16* base, int ld) {
  int lane = threadIdx.x & 31;
  const __bf16* p = base + (size_t)(lane & 15) * ld + ((lane >> 4) << 3);
  union { v16bf v; uint4 q[2]; } u;
  u.q[0] = *(const uint4*)(p);
  u.q[1] = *(const uint4*)(p + 16);
  return u.v;
}
// B (32x16 bf16, KxN) loaded from B^T stored row-major [N][K]:
// lane<16: N=lane, K=0..7 / K=16..23; lane>=16: K=8..15 / K=24..31.
static __device__ __forceinline__ v16bf load_frag_b(const __bf16* base, int ld) {
  int lane = threadIdx.x & 31;
  const __bf16* p = base + (size_t)(lane & 15) * ld + ((lane >> 4) << 3);
  union { v16bf v; uint4 q[2]; } u;
  u.q[0] = *(const uint4*)(p);
  u.q[1] = *(const uint4*)(p + 16);
  return u.v;
}
static __device__ __forceinline__ v8f wmma_bf(v16bf a, v16bf b, v8f c) {
  return __builtin_amdgcn_wmma_f32_16x16x32_bf16(false, a, false, b, (short)0, c, false, false);
}

// ---------- weight transpose + fp32->bf16 : dst[n*K+k] = src[k*N+n] ----------
__global__ void tconv_kernel(const float* __restrict__ src, __bf16* __restrict__ dst,
                             int K, int N) {
  size_t total = (size_t)K * N;
  for (size_t i = (size_t)blockIdx.x * blockDim.x + threadIdx.x; i < total;
       i += (size_t)gridDim.x * blockDim.x) {
    size_t n = i / (size_t)K, k = i % (size_t)K;
    dst[i] = f2bf(src[k * (size_t)N + n]);
  }
}

// ---------- LayerNorm (fp32 in -> bf16 out), one row per block ----------
__global__ __launch_bounds__(256) void ln_kernel(const float* __restrict__ x,
                          const float* __restrict__ gamma,
                          const float* __restrict__ beta,
                          __bf16* __restrict__ out) {
  int row = blockIdx.x;
  const float* xr = x + (size_t)row * DD;
  __shared__ float rs[8], rs2[8];
  float s = 0.f, s2 = 0.f;
  for (int i = threadIdx.x; i < DD; i += 256) { float v = xr[i]; s += v; s2 += v * v; }
  for (int o = 16; o; o >>= 1) { s += __shfl_xor(s, o, 32); s2 += __shfl_xor(s2, o, 32); }
  int wid = threadIdx.x >> 5, ln = threadIdx.x & 31;
  if (ln == 0) { rs[wid] = s; rs2[wid] = s2; }
  __syncthreads();
  if (threadIdx.x == 0) {
    float a = 0.f, b = 0.f;
    for (int i = 0; i < 8; ++i) { a += rs[i]; b += rs2[i]; }
    rs[0] = a; rs2[0] = b;
  }
  __syncthreads();
  float mean = rs[0] * (1.0f / DD);
  float var  = rs2[0] * (1.0f / DD) - mean * mean;
  float inv  = rsqrtf(var + 1e-5f);
  for (int i = threadIdx.x; i < DD; i += 256)
    out[(size_t)row * DD + i] = f2bf((xr[i] - mean) * inv * gamma[i] + beta[i]);
}

// ---------- WMMA GEMM: C[M,N] = A[M,K](bf16) x Bt[N,K](bf16)^T + bias ----------
// EPI 0: store bf16(acc+bias); EPI 1: store bf16(gelu(acc+bias));
// EPI 2: store f32 res + ls*(acc+bias)
template <int EPI>
__global__ __launch_bounds__(256) void gemm_wmma(const __bf16* __restrict__ A,
                          const __bf16* __restrict__ Bt,
                          const float* __restrict__ bias,
                          const float* __restrict__ res,
                          const float* __restrict__ ls,
                          float* __restrict__ outF, __bf16* __restrict__ outH,
                          int M, int N, int K) {
  int wave = blockIdx.x * (blockDim.x >> 5) + (threadIdx.x >> 5);
  int tilesN = N >> 6;
  int tm = wave / tilesN, tn = wave % tilesN;
  int m0 = tm * 32, n0 = tn * 64;
  if (m0 >= M) return;
  v8f z = {0.f, 0.f, 0.f, 0.f, 0.f, 0.f, 0.f, 0.f};
  v8f c[2][4];
#pragma unroll
  for (int i = 0; i < 2; ++i)
#pragma unroll
    for (int j = 0; j < 4; ++j) c[i][j] = z;

  for (int k0 = 0; k0 < K; k0 += 32) {
    __builtin_prefetch((const void*)(A + (size_t)m0 * K + k0 + 128), 0, 0);
    v16bf a0 = load_frag_a(A + (size_t)m0 * K + k0, K);
    v16bf a1 = load_frag_a(A + (size_t)(m0 + 16) * K + k0, K);
#pragma unroll
    for (int j = 0; j < 4; ++j) {
      v16bf b = load_frag_b(Bt + (size_t)(n0 + j * 16) * K + k0, K);
      c[0][j] = wmma_bf(a0, b, c[0][j]);
      c[1][j] = wmma_bf(a1, b, c[1][j]);
    }
  }
  int lane = threadIdx.x & 31;
  int hi = lane >> 4, nn = lane & 15;
#pragma unroll
  for (int i = 0; i < 2; ++i) {
#pragma unroll
    for (int j = 0; j < 4; ++j) {
#pragma unroll
      for (int r = 0; r < 8; ++r) {
        int row = m0 + i * 16 + r + hi * 8;
        int col = n0 + j * 16 + nn;
        float acc = c[i][j][r] + bias[col];
        size_t idx = (size_t)row * N + col;
        if (EPI == 0) {
          outH[idx] = f2bf(acc);
        } else if (EPI == 1) {
          float g = 0.5f * acc * (1.0f + erff(acc * 0.70710678118654752f));
          outH[idx] = f2bf(g);
        } else {
          outF[idx] = res[idx] + ls[col] * acc;
        }
      }
    }
  }
}

// ---------- RoPE + head rearrange: qkv[BS,3D] -> q,k [B,H,S,64], v^T [B,H,64,S] ----------
__global__ void rope_rearrange(const __bf16* __restrict__ qkv,
                               __bf16* __restrict__ qh, __bf16* __restrict__ kh,
                               __bf16* __restrict__ vT) {
  int idx = blockIdx.x * blockDim.x + threadIdx.x;  // B*H*S*DHALF = 2^22
  if (idx >= BB * HH * SS * DHALF) return;
  int d = idx & (DHALF - 1);
  int t = idx >> 5;
  int s = t & (SS - 1); t >>= 9;
  int h = t & (HH - 1);
  int b = t >> 4;
  float freq = __expf(-(float)d * (9.210340371976184f / (float)DHALF));  // ln(10000)
  float ang = (float)s * freq;
  float sn, cs;
  __sincosf(ang, &sn, &cs);
  size_t base = ((size_t)(b * SS + s)) * (3 * DD);
  int c0 = h * HDIM + d;
  float q1 = bf2f(qkv[base + c0]),          q2 = bf2f(qkv[base + c0 + DHALF]);
  float k1 = bf2f(qkv[base + DD + c0]),     k2 = bf2f(qkv[base + DD + c0 + DHALF]);
  float v1 = bf2f(qkv[base + 2 * DD + c0]), v2 = bf2f(qkv[base + 2 * DD + c0 + DHALF]);
  size_t ho = ((size_t)(b * HH + h) * SS + s) * HDIM;
  qh[ho + d]         = f2bf(q1 * cs - q2 * sn);
  qh[ho + d + DHALF] = f2bf(q1 * sn + q2 * cs);
  kh[ho + d]         = f2bf(k1 * cs - k2 * sn);
  kh[ho + d + DHALF] = f2bf(k1 * sn + k2 * cs);
  size_t vo = (size_t)(b * HH + h) * HDIM;
  vT[(vo + d) * SS + s]         = f2bf(v1);
  vT[(vo + d + DHALF) * SS + s] = f2bf(v2);
}

// ---------- attention: one wave per (b,h, 16 query rows) ----------
__global__ __launch_bounds__(32) void attention_kernel(const __bf16* __restrict__ qh,
                                 const __bf16* __restrict__ kh,
                                 const __bf16* __restrict__ vT,
                                 __bf16* __restrict__ outp) {
  __shared__ float sc[16][SS + 4];
  __shared__ float rsum[16];
  int bh = blockIdx.x;          // b*H + h
  int q0 = blockIdx.y * 16;
  int b = bh / HH, h = bh % HH;
  int lane = threadIdx.x;
  int hi = lane >> 4, nn = lane & 15;

  const __bf16* qbase = qh + ((size_t)bh * SS + q0) * HDIM;
  v16bf aq0 = load_frag_a(qbase, HDIM);        // hd 0..31
  v16bf aq1 = load_frag_a(qbase + 32, HDIM);   // hd 32..63
  v8f z = {0.f, 0.f, 0.f, 0.f, 0.f, 0.f, 0.f, 0.f};

  // scores = (q @ k^T) / 8  -> LDS strip [16][512]
  for (int kt = 0; kt < SS / 16; ++kt) {
    const __bf16* kbase = kh + ((size_t)bh * SS + kt * 16) * HDIM;
    v16bf bk0 = load_frag_b(kbase, HDIM);
    v16bf bk1 = load_frag_b(kbase + 32, HDIM);
    v8f cc = z;
    cc = wmma_bf(aq0, bk0, cc);
    cc = wmma_bf(aq1, bk1, cc);
#pragma unroll
    for (int r = 0; r < 8; ++r)
      sc[r + hi * 8][kt * 16 + nn] = cc[r] * 0.125f;
  }
  __syncthreads();

  // row-wise softmax numerator (keep 1/sum for epilogue)
  for (int r = 0; r < 16; ++r) {
    float mx = -1e30f;
    for (int i = lane; i < SS; i += 32) mx = fmaxf(mx, sc[r][i]);
    for (int o2 = 16; o2; o2 >>= 1) mx = fmaxf(mx, __shfl_xor(mx, o2, 32));
    float sum = 0.f;
    for (int i = lane; i < SS; i += 32) {
      float e = __expf(sc[r][i] - mx);
      sc[r][i] = e;
      sum += e;
    }
    for (int o2 = 16; o2; o2 >>= 1) sum += __shfl_xor(sum, o2, 32);
    if (lane == 0) rsum[r] = sum;
  }
  __syncthreads();

  // o = P @ V  (contraction over 512 keys in 32-chunks)
  v8f oacc[4];
#pragma unroll
  for (int j = 0; j < 4; ++j) oacc[j] = z;
  for (int st = 0; st < SS / 32; ++st) {
    union { v16bf v; unsigned short s[16]; } ua;
    int kb = st * 32 + hi * 8;
#pragma unroll
    for (int e = 0; e < 16; ++e) {
      int k = kb + (e < 8 ? e : e + 8);
      ua.s[e] = f2bf_bits(sc[nn][k]);
    }
#pragma unroll
    for (int j = 0; j < 4; ++j) {
      const __bf16* vb = vT + ((size_t)bh * HDIM + j * 16) * SS + st * 32;
      v16bf bv = load_frag_b(vb, SS);
      oacc[j] = wmma_bf(ua.v, bv, oacc[j]);
    }
  }
#pragma unroll
  for (int j = 0; j < 4; ++j) {
#pragma unroll
    for (int r = 0; r < 8; ++r) {
      int m = r + hi * 8;
      float inv = 1.0f / rsum[m];
      size_t row = (size_t)b * SS + q0 + m;
      outp[row * DD + h * HDIM + j * 16 + nn] = f2bf(oacc[j][r] * inv);
    }
  }
}

// ---------------------------------------------------------------------------
extern "C" void kernel_launch(void* const* d_in, const int* in_sizes, int n_in,
                              void* d_out, int out_size, void* d_ws, size_t ws_size,
                              hipStream_t stream) {
  const float* src  = (const float*)d_in[0];
  const float* wqkv = (const float*)d_in[1];
  const float* bqkv = (const float*)d_in[2];
  const float* wo   = (const float*)d_in[3];
  const float* bo   = (const float*)d_in[4];
  const float* g1   = (const float*)d_in[5];
  const float* be1  = (const float*)d_in[6];
  const float* g2   = (const float*)d_in[7];
  const float* be2  = (const float*)d_in[8];
  const float* ls1  = (const float*)d_in[9];
  const float* ls2  = (const float*)d_in[10];
  const float* w1   = (const float*)d_in[11];
  const float* bf1  = (const float*)d_in[12];
  const float* w2   = (const float*)d_in[13];
  const float* bf2  = (const float*)d_in[14];
  float* out = (float*)d_out;

  char* ws = (char*)d_ws;
  // weights (bf16, transposed [N][K])
  __bf16* wqkvT = (__bf16*)(ws + 0);           // 3072*1024*2 = 6.0 MiB
  __bf16* woT   = (__bf16*)(ws + 6291456);     // 1024*1024*2 = 2.0 MiB
  __bf16* w1T   = (__bf16*)(ws + 8388608);     // 4096*1024*2 = 8.0 MiB
  __bf16* w2T   = (__bf16*)(ws + 16777216);    // 1024*4096*2 = 8.0 MiB
  // region R1 (16 MiB): h1 (LN1 out) then attention output o (h1 dead by then)
  __bf16* h1    = (__bf16*)(ws + 25165824);
  __bf16* obuf  = h1;
  // region R2 (48 MiB): qkv bf16, then x f32 (qkv dead after rope_rearrange)
  __bf16* qkv   = (__bf16*)(ws + 41943040);
  float*  xbuf  = (float*)(ws + 41943040);
  // region R3 (48 MiB): q/k/vT for attention; h2 reuses q slot afterwards
  __bf16* qh    = (__bf16*)(ws + 92274688);
  __bf16* kh    = (__bf16*)(ws + 109051904);
  __bf16* vT    = (__bf16*)(ws + 125829120);
  __bf16* h2    = qh;
  // f1 (64 MiB), end of ws usage at 200 MiB
  __bf16* f1    = (__bf16*)(ws + 142606336);

  // 1) weight convert+transpose
  tconv_kernel<<<2048, 256, 0, stream>>>(wqkv, wqkvT, DD, 3 * DD);
  tconv_kernel<<<2048, 256, 0, stream>>>(wo,   woT,   DD, DD);
  tconv_kernel<<<2048, 256, 0, stream>>>(w1,   w1T,   DD, DFFN);
  tconv_kernel<<<2048, 256, 0, stream>>>(w2,   w2T,   DFFN, DD);

  // 2) LN1
  ln_kernel<<<BSROWS, 256, 0, stream>>>(src, g1, be1, h1);

  // 3) QKV GEMM: [8192,1024] x [1024,3072]   -> 12288 wave tiles / 8 = 1536 blocks
  gemm_wmma<0><<<1536, 256, 0, stream>>>(h1, wqkvT, bqkv, nullptr, nullptr,
                                         nullptr, qkv, BSROWS, 3 * DD, DD);

  // 4) RoPE + head split
  rope_rearrange<<<(BB * HH * SS * DHALF) / 256, 256, 0, stream>>>(qkv, qh, kh, vT);

  // 5) attention
  dim3 ag(BB * HH, SS / 16);
  attention_kernel<<<ag, 32, 0, stream>>>(qh, kh, vT, obuf);

  // 6) Wo GEMM + residual1: x = src + ls1*(o@wo + bo)
  gemm_wmma<2><<<512, 256, 0, stream>>>(obuf, woT, bo, src, ls1, xbuf, nullptr,
                                        BSROWS, DD, DD);

  // 7) LN2
  ln_kernel<<<BSROWS, 256, 0, stream>>>(xbuf, g2, be2, h2);

  // 8) FFN1 + exact GELU
  gemm_wmma<1><<<2048, 256, 0, stream>>>(h2, w1T, bf1, nullptr, nullptr,
                                         nullptr, f1, BSROWS, DFFN, DD);

  // 9) FFN2 + residual2 -> out
  gemm_wmma<2><<<512, 256, 0, stream>>>(f1, w2T, bf2, xbuf, ls2, out, nullptr,
                                        BSROWS, DD, DFFN);
}
